// AdvancedRPN_23854248362779
// MI455X (gfx1250) — matrix operand order repaired and checked by reading, not compiled, not run on hardware
//
#include <hip/hip_runtime.h>
#include <hip/hip_bf16.h>
#include <math.h>

typedef _Float16 f16;
typedef __attribute__((ext_vector_type(16))) _Float16 v16h;
typedef __attribute__((ext_vector_type(8)))  _Float16 v8h;
typedef __attribute__((ext_vector_type(8)))  float    v8f;

#define RPN_FH   200
#define RPN_FW   200
#define NPIX     40000          // 200*200
#define CIN      256
#define COUT     256
#define KTOT     2304           // 9*256
#define NANCH    120000         // NPIX*3
#define NSORT    131072         // next pow2
#define TOPK     1000
#define NMS_T    0.7f
#define MIN_SZ   0.001f
#define BBOX_CLIP 4.135166556742356f   // log(1000/16)

// ---------------------------------------------------------------------------
// 1) features f32 NCHW -> f16 NHWC  (xh[p*256 + c])
// ---------------------------------------------------------------------------
__global__ __launch_bounds__(256)
void cvt_feat(const float* __restrict__ x, f16* __restrict__ xh) {
    int p  = blockIdx.x * 256 + threadIdx.x;
    int c0 = blockIdx.y * 8;
    if (p >= NPIX) return;
    v8h v;
    #pragma unroll
    for (int q = 0; q < 8; ++q)
        v[q] = (f16)x[(size_t)(c0 + q) * NPIX + p];
    *(v8h*)(xh + (size_t)p * CIN + c0) = v;
}

// ---------------------------------------------------------------------------
// 2) conv_w [o][ci][ky][kx] f32 -> f16 B-matrix wh[o][ (ky*3+kx)*256 + ci ]
// ---------------------------------------------------------------------------
__global__ __launch_bounds__(256)
void cvt_w(const float* __restrict__ w, f16* __restrict__ wh) {
    int t = blockIdx.x * 256 + threadIdx.x;
    if (t >= COUT * CIN * 9) return;
    int kxy = t % 9;
    int ci  = (t / 9) % CIN;
    int o   = t / (9 * CIN);
    wh[(size_t)o * KTOT + kxy * CIN + ci] = (f16)w[t];
}

// ---------------------------------------------------------------------------
// 3) 3x3 conv as implicit GEMM with WMMA.
//    M = 40000 pixels, N = 256 out-ch, K = 2304.
//    Block: 64x64 tile, 128 threads (4 waves), wave -> 2x2 16x16 WMMA tiles.
//    K-step = 32 = one (ky,kx) tap, one 32-channel chunk (contiguous in NHWC).
// ---------------------------------------------------------------------------
__global__ __launch_bounds__(128)
void conv3x3_wmma(const f16* __restrict__ xh, const f16* __restrict__ wh,
                  const float* __restrict__ bias, f16* __restrict__ th) {
    __shared__ __attribute__((aligned(64))) f16 As[64 * 32];
    __shared__ __attribute__((aligned(64))) f16 Bs[64 * 32];

    const int tid  = threadIdx.x;
    const int lane = tid & 31;
    const int wave = tid >> 5;
    const int wr   = wave >> 1;   // 0..1  (M half)
    const int wc   = wave & 1;    // 0..1  (N half)
    const int m0   = blockIdx.x * 64;   // pixel base (625 blocks, exact)
    const int n0   = blockIdx.y * 64;   // out-channel base (4 blocks)

    v8f acc[2][2] = {};

    // cooperative tile-load assignment: 128 threads, 64 rows x 2 chunks(16 halves)
    const int arow = tid >> 1;
    const int achk = tid & 1;
    const int apix = m0 + arow;
    const int ay   = apix / RPN_FW;
    const int ax   = apix % RPN_FW;

    for (int s = 0; s < 72; ++s) {
        const int kk  = s * 32;
        const int pos = s >> 3;          // 0..8 tap index
        const int ky  = pos / 3 - 1;
        const int kx  = pos % 3 - 1;
        const int ci0 = (s & 7) * 32;

        // A tile: 64 pixels x 32 halves of this tap/channel chunk (zero-padded)
        {
            int yy = ay + ky, xx = ax + kx;
            v16h v = {};
            if (yy >= 0 && yy < RPN_FH && xx >= 0 && xx < RPN_FW)
                v = *(const v16h*)(xh + ((size_t)(yy * RPN_FW + xx) * CIN) + ci0 + achk * 16);
            *(v16h*)(As + arow * 32 + achk * 16) = v;
        }
        // B tile: 64 out-channels x 32 k (contiguous in wh)
        {
            v16h v = *(const v16h*)(wh + (size_t)(n0 + arow) * KTOT + kk + achk * 16);
            *(v16h*)(Bs + arow * 32 + achk * 16) = v;
        }
        __syncthreads();

        // fragments: contiguous k order (k = 16*(lane/16) + s), same for A and B
        v16h afrag[2], bfrag[2];
        const int hs = (lane >> 4) * 16;  // A: lanes 0-15 -> K 0..15, 16-31 -> K 16..31
        #pragma unroll
        for (int mt = 0; mt < 2; ++mt) {
            const int r = wr * 32 + mt * 16 + (lane & 15);
            afrag[mt] = *(const v16h*)(As + r * 32 + hs);
        }
        const int ks = (lane >> 4) * 16;  // B: lanes 0-15 -> K 0..15, 16-31 -> K 16..31
        #pragma unroll
        for (int nt = 0; nt < 2; ++nt) {
            const int r = wc * 32 + nt * 16 + (lane & 15);
            bfrag[nt] = *(const v16h*)(Bs + r * 32 + ks);
        }
        #pragma unroll
        for (int mt = 0; mt < 2; ++mt)
            #pragma unroll
            for (int nt = 0; nt < 2; ++nt)
                acc[mt][nt] = __builtin_amdgcn_wmma_f32_16x16x32_f16(
                    false, afrag[mt], false, bfrag[nt],
                    (short)0, acc[mt][nt], false, false);
        __syncthreads();
    }

    // epilogue: bias + relu, store f16 NHWC. C layout: M = r + 8*(lane>=16), N = lane%16
    #pragma unroll
    for (int mt = 0; mt < 2; ++mt) {
        #pragma unroll
        for (int nt = 0; nt < 2; ++nt) {
            const int n = n0 + wc * 32 + nt * 16 + (lane & 15);
            const float b = bias[n];
            const int mbase = m0 + wr * 32 + mt * 16 + (lane >> 4) * 8;
            #pragma unroll
            for (int r = 0; r < 8; ++r) {
                float v = acc[mt][nt][r] + b;
                v = v > 0.f ? v : 0.f;
                th[(size_t)(mbase + r) * COUT + n] = (f16)v;
            }
        }
    }
}

// ---------------------------------------------------------------------------
// 4) 1x1 heads + anchor decode + clip + sort-key build
// ---------------------------------------------------------------------------
__global__ __launch_bounds__(256)
void heads_decode(const f16* __restrict__ th,
                  const float* __restrict__ cls_w, const float* __restrict__ cls_b,
                  const float* __restrict__ bbox_w, const float* __restrict__ bbox_b,
                  const int* __restrict__ ih_p, const int* __restrict__ iw_p,
                  float* __restrict__ boxes, float* __restrict__ scores,
                  unsigned long long* __restrict__ keys) {
    __shared__ float wl[15 * 256];
    for (int i = threadIdx.x; i < 15 * 256; i += 256)
        wl[i] = (i < 3 * 256) ? cls_w[i] : bbox_w[i - 3 * 256];
    __syncthreads();

    const int p = blockIdx.x * 256 + threadIdx.x;
    if (p >= NPIX) return;

    float acc[15];
    #pragma unroll
    for (int k = 0; k < 15; ++k) acc[k] = 0.f;
    const f16* row = th + (size_t)p * CIN;
    for (int c8 = 0; c8 < 32; ++c8) {
        v8h v = *(const v8h*)(row + c8 * 8);
        #pragma unroll
        for (int q = 0; q < 8; ++q) {
            float vv = (float)v[q];
            int c = c8 * 8 + q;
            #pragma unroll
            for (int k = 0; k < 15; ++k) acc[k] += vv * wl[k * 256 + c];
        }
    }

    const int image_h = *ih_p, image_w = *iw_p;
    const float cw = (float)image_w, ch = (float)image_h;
    const int sx = image_w / RPN_FW, sy = image_h / RPN_FH;
    const float shx = (float)((p % RPN_FW) * sx);
    const float shy = (float)((p / RPN_FW) * sy);
    // cell anchors (scale 32, ratios .5/1/2, rounded): widths/heights
    const float AW[3] = {46.f, 32.f, 22.f};
    const float AH[3] = {22.f, 32.f, 46.f};

    #pragma unroll
    for (int a = 0; a < 3; ++a) {
        const int i = p * 3 + a;
        float sc = acc[a] + cls_b[a];
        float dx = acc[3 + a * 4 + 0] + bbox_b[a * 4 + 0];
        float dy = acc[3 + a * 4 + 1] + bbox_b[a * 4 + 1];
        float dw = fminf(acc[3 + a * 4 + 2] + bbox_b[a * 4 + 2], BBOX_CLIP);
        float dh = fminf(acc[3 + a * 4 + 3] + bbox_b[a * 4 + 3], BBOX_CLIP);
        float w = AW[a], h = AH[a];
        float pcx = dx * w + shx, pcy = dy * h + shy;
        float pw = expf(dw) * w, ph = expf(dh) * h;
        float x1 = fminf(fmaxf(pcx - 0.5f * pw, 0.f), cw);
        float y1 = fminf(fmaxf(pcy - 0.5f * ph, 0.f), ch);
        float x2 = fminf(fmaxf(pcx + 0.5f * pw, 0.f), cw);
        float y2 = fminf(fmaxf(pcy + 0.5f * ph, 0.f), ch);
        boxes[(size_t)i * 4 + 0] = x1;
        boxes[(size_t)i * 4 + 1] = y1;
        boxes[(size_t)i * 4 + 2] = x2;
        boxes[(size_t)i * 4 + 3] = y2;
        scores[i] = sc;
        unsigned u = __float_as_uint(sc);
        unsigned ord = (u & 0x80000000u) ? ~u : (u | 0x80000000u);
        // ascending sort => descending score, ties -> smaller index first
        keys[i] = (((unsigned long long)(~ord)) << 32) | (unsigned)i;
    }
}

__global__ __launch_bounds__(256)
void pad_keys(unsigned long long* __restrict__ keys) {
    int i = NANCH + blockIdx.x * 256 + threadIdx.x;
    if (i < NSORT) keys[i] = ~0ull;
}

// ---------------------------------------------------------------------------
// 5) bitonic sort step (ascending), fixed 153-launch schedule
// ---------------------------------------------------------------------------
__global__ __launch_bounds__(256)
void bitonic_step(unsigned long long* __restrict__ d, int j, int k) {
    unsigned i = blockIdx.x * 256 + threadIdx.x;
    unsigned ixj = i ^ (unsigned)j;
    if (ixj > i) {
        unsigned long long a = d[i], b = d[ixj];
        bool up = ((i & (unsigned)k) == 0);
        if ((a > b) == up) { d[i] = b; d[ixj] = a; }
    }
}

// ---------------------------------------------------------------------------
// 6) greedy NMS on top-1000 (single workgroup), write (out[1000x5], keep[1000])
// ---------------------------------------------------------------------------
__global__ __launch_bounds__(1024)
void nms_out(const unsigned long long* __restrict__ keys,
             const float* __restrict__ boxes, const float* __restrict__ scores,
             float* __restrict__ out) {
    __shared__ float bx1[TOPK], by1[TOPK], bx2[TOPK], by2[TOPK], bar[TOPK], bsc[TOPK];
    __shared__ int   keep[TOPK];
    const int j = threadIdx.x;
    if (j < TOPK) {
        unsigned long long key = keys[j];
        int idx = (int)(unsigned)(key & 0xFFFFFFFFull);
        float x1 = boxes[(size_t)idx * 4 + 0];
        float y1 = boxes[(size_t)idx * 4 + 1];
        float x2 = boxes[(size_t)idx * 4 + 2];
        float y2 = boxes[(size_t)idx * 4 + 3];
        bx1[j] = x1; by1[j] = y1; bx2[j] = x2; by2[j] = y2;
        bar[j] = (x2 - x1) * (y2 - y1);
        bsc[j] = scores[idx];
        keep[j] = ((x2 - x1) >= MIN_SZ) && ((y2 - y1) >= MIN_SZ);
    }
    __syncthreads();
    for (int i = 0; i < TOPK; ++i) {
        if (j < TOPK && j > i && keep[j] && keep[i]) {
            float xl = fmaxf(bx1[i], bx1[j]);
            float yt = fmaxf(by1[i], by1[j]);
            float xr = fminf(bx2[i], bx2[j]);
            float yb = fminf(by2[i], by2[j]);
            float inter = fmaxf(xr - xl, 0.f) * fmaxf(yb - yt, 0.f);
            float iou = inter / (bar[i] + bar[j] - inter + 1e-8f);
            if (iou > NMS_T) keep[j] = 0;
        }
        __syncthreads();
    }
    if (j < TOPK) {
        bool k = keep[j] != 0;
        out[j * 5 + 0] = k ? bx1[j] : 0.f;
        out[j * 5 + 1] = k ? by1[j] : 0.f;
        out[j * 5 + 2] = k ? bx2[j] : 0.f;
        out[j * 5 + 3] = k ? by2[j] : 0.f;
        out[j * 5 + 4] = k ? bsc[j] : 0.f;
        out[TOPK * 5 + j] = k ? 1.f : 0.f;
    }
}

// ---------------------------------------------------------------------------
extern "C" void kernel_launch(void* const* d_in, const int* in_sizes, int n_in,
                              void* d_out, int out_size, void* d_ws, size_t ws_size,
                              hipStream_t stream) {
    const float* features = (const float*)d_in[0];
    const float* conv_w   = (const float*)d_in[1];
    const float* conv_b   = (const float*)d_in[2];
    const float* cls_w    = (const float*)d_in[3];
    const float* cls_b    = (const float*)d_in[4];
    const float* bbox_w   = (const float*)d_in[5];
    const float* bbox_b   = (const float*)d_in[6];
    const int*   image_h  = (const int*)d_in[7];
    const int*   image_w  = (const int*)d_in[8];
    float* out = (float*)d_out;

    // workspace carve-out (256B aligned)
    char* base = (char*)d_ws;
    size_t off = 0;
    auto carve = [&](size_t bytes) {
        char* p = base + off;
        off = (off + bytes + 255) & ~(size_t)255;
        return p;
    };
    f16*   xh     = (f16*)carve((size_t)NPIX * CIN * sizeof(f16));     // ~19.5 MB
    f16*   wh     = (f16*)carve((size_t)COUT * KTOT * sizeof(f16));    // ~1.1 MB
    f16*   th     = (f16*)carve((size_t)NPIX * COUT * sizeof(f16));    // ~19.5 MB
    float* boxes  = (float*)carve((size_t)NANCH * 4 * sizeof(float));  // ~1.9 MB
    float* scores = (float*)carve((size_t)NANCH * sizeof(float));      // ~0.5 MB
    unsigned long long* keys = (unsigned long long*)carve((size_t)NSORT * 8); // 1 MB

    // 1) layout conversions to f16
    cvt_feat<<<dim3((NPIX + 255) / 256, CIN / 8), 256, 0, stream>>>(features, xh);
    cvt_w<<<(COUT * CIN * 9 + 255) / 256, 256, 0, stream>>>(conv_w, wh);

    // 2) 3x3 conv + relu via WMMA implicit GEMM
    conv3x3_wmma<<<dim3(NPIX / 64, COUT / 64), 128, 0, stream>>>(xh, wh, conv_b, th);

    // 3) heads + decode + sort keys
    heads_decode<<<(NPIX + 255) / 256, 256, 0, stream>>>(
        th, cls_w, cls_b, bbox_w, bbox_b, image_h, image_w, boxes, scores, keys);
    pad_keys<<<(NSORT - NANCH + 255) / 256, 256, 0, stream>>>(keys);

    // 4) full bitonic sort of 2^17 keys (ascending) => top-K in keys[0..999]
    for (int k = 2; k <= NSORT; k <<= 1)
        for (int j = k >> 1; j > 0; j >>= 1)
            bitonic_step<<<NSORT / 256, 256, 0, stream>>>(keys, j, k);

    // 5) NMS + final outputs
    nms_out<<<1, 1024, 0, stream>>>(keys, boxes, scores, out);
}
